// MyRNNCell_15831249453423
// MI455X (gfx1250) — matrix operand order repaired and checked by reading, not compiled, not run on hardware
//
#include <hip/hip_runtime.h>
#include <cmath>

typedef __attribute__((ext_vector_type(16))) __bf16 v16bf;
typedef __attribute__((ext_vector_type(8)))  __bf16 v8bf;
typedef __attribute__((ext_vector_type(4)))  __bf16 v4bf;
typedef __attribute__((ext_vector_type(8)))  float  v8f;
typedef __attribute__((ext_vector_type(4)))  float  v4f;

#define B_  256
#define T_  1024
#define D_  128
#define U_  256
#define DU_ 384
#define RS  392   // padded LDS row stride in bf16 elems (784 B, 16B-aligned, bank-friendly)

#if __has_builtin(__builtin_amdgcn_tanhf)
__device__ __forceinline__ float fast_tanh(float x) { return __builtin_amdgcn_tanhf(x); }
#else
__device__ __forceinline__ float fast_tanh(float x) {
  return 1.0f - 2.0f / (__expf(2.0f * x) + 1.0f);
}
#endif

struct StepCtx {
  const float* __restrict__ x;
  float* __restrict__ out;
  long  xrow;       // global batch row this wave streams
  int   d0;         // lane's x column base
  int   nl, half, n0, b0, wave;
  float bv;
};

// A 16x32 bf16 fragment: lane m = lane&15; elems 0..7 = K[k0+half*8 ..+7],
// elems 8..15 = K[k0+16+half*8 ..+7]  (two contiguous 16B LDS chunks).
__device__ __forceinline__ v16bf load_a(const __bf16* __restrict__ arow, int k0) {
  v8bf c1 = *(const v8bf*)(arow + k0);
  v8bf c2 = *(const v8bf*)(arow + k0 + 16);
  return __builtin_shufflevector(c1, c2, 0,1,2,3,4,5,6,7,8,9,10,11,12,13,14,15);
}

// One RNN timestep: reads [h|x_t] from `cur`, writes h_t into `nxt` h-region,
// stages x_{t+1} regs, writes f32 output. Exactly one barrier.
__device__ __forceinline__ void rnn_step(const StepCtx& c, int t,
                                         __bf16* __restrict__ cur,
                                         __bf16* __restrict__ nxt,
                                         v4f& xr, const v16bf* wreg)
{
  // issue x_{t+1} prefetch FIRST (independent of LDS; barrier fence pins it early,
  // giving a full compute phase of latency slack). Non-temporal: read-once stream.
  const int tn = (t + 1 < T_) ? (t + 1) : t;
  const v4f* xp = (const v4f*)(c.x + (c.xrow * T_ + tn) * D_ + c.d0);
  v4f xrn = __builtin_nontemporal_load(xp);

  // stage x_t (bf16) into cur x-region (cols 256..383), row = wave
  {
    v4bf xc;
    xc[0] = (__bf16)xr[0]; xc[1] = (__bf16)xr[1];
    xc[2] = (__bf16)xr[2]; xc[3] = (__bf16)xr[3];
    *(v4bf*)(&cur[c.wave * RS + U_ + c.d0]) = xc;
  }
  __syncthreads();   // x_t staged + h_{t-1} (written pre-barrier by all waves) visible

  // acc = bias + [h | x_t](16x384) @ Wslice(384x16)
  // dual accumulators break the WMMA RAW chain; A-loads software-pipelined one
  // stage ahead so each WMMA needs only dscnt<=2 (LDS latency hides behind WMMA).
  v8f acc0 = { c.bv, c.bv, c.bv, c.bv, c.bv, c.bv, c.bv, c.bv };
  v8f acc1 = { 0.f, 0.f, 0.f, 0.f, 0.f, 0.f, 0.f, 0.f };
  const __bf16* arow = &cur[c.nl * RS + c.half * 8];

  v16bf a_cur = load_a(arow, 0);
  #pragma unroll
  for (int kk = 0; kk < 12; ++kk) {
    v16bf a_nxt = (kk < 11) ? load_a(arow, (kk + 1) * 32) : a_cur;
    if (kk & 1)
      acc1 = __builtin_amdgcn_wmma_f32_16x16x32_bf16(false, a_cur, false, wreg[kk],
                                                     (short)0, acc1, false, false);
    else
      acc0 = __builtin_amdgcn_wmma_f32_16x16x32_bf16(false, a_cur, false, wreg[kk],
                                                     (short)0, acc0, false, false);
    a_cur = a_nxt;
  }

  // tanh (hardware TRANS op), write h_t to nxt, stream out f32
  // C/D layout: lane n = lane&15, VGPR r -> row m = r + 8*half
  #pragma unroll
  for (int r = 0; r < 8; ++r) {
    const int m = r + c.half * 8;
    const float v = fast_tanh(acc0[r] + acc1[r]);
    nxt[m * RS + c.n0 + c.nl] = (__bf16)v;
    __builtin_nontemporal_store(
        v, &c.out[(((long)(c.b0 + m)) * T_ + t) * U_ + c.n0 + c.nl]);
  }

  xr = xrn;
}

__global__ __launch_bounds__(512)
void rnn_scan_wmma(const float* __restrict__ x,
                   const float* __restrict__ h0,
                   const float* __restrict__ W,
                   const float* __restrict__ bias,
                   float* __restrict__ out)
{
  // ping-pong [h (0..255) | x_t (256..383)] bf16 buffers, 16 batch rows each
  __shared__ __bf16 smem[2 * 16 * RS];
  __bf16* buf0 = smem;
  __bf16* buf1 = smem + 16 * RS;

  const int tid  = threadIdx.x;
  const int wave = tid >> 5;      // 0..15 : N-tile owner
  const int lane = tid & 31;

  StepCtx c;
  c.x = x; c.out = out;
  c.wave = wave;
  c.nl   = lane & 15;
  c.half = lane >> 4;
  c.n0   = wave * 16;
  c.b0   = blockIdx.x * 16;
  c.xrow = c.b0 + wave;
  c.d0   = lane * 4;

  // ---- W slice (384 x 16 cols n0..n0+15) resident in registers, WMMA B layout.
  // B 32x16 bf16: lane n = lane&15; element j<8 holds K = k0 + half*8 + j, j>=8 holds K = k0 + 16 + half*8 + (j-8).
  v16bf wreg[12];
  #pragma unroll
  for (int kk = 0; kk < 12; ++kk) {
    const int k0 = kk * 32;
    v16bf wf;
    #pragma unroll
    for (int j = 0; j < 16; ++j) {
      const int k = k0 + c.half * 8 + j + ((j >= 8) ? 8 : 0);
      wf[j] = (__bf16)W[k * U_ + c.n0 + c.nl];
    }
    wreg[kk] = wf;
  }

  c.bv = bias[c.n0 + c.nl];

  // ---- init buf0 h-region from h0 (bf16)
  {
    const int r  = tid >> 5;         // 0..15
    const int c0 = (tid & 31) * 8;   // 0..248
    #pragma unroll
    for (int i = 0; i < 8; ++i)
      buf0[r * RS + c0 + i] = (__bf16)h0[(c.b0 + r) * U_ + c0 + i];
  }

  // ---- preload x(t=0): wave w streams batch row (b0+w); lane covers 4 floats
  v4f xr = __builtin_nontemporal_load(
      (const v4f*)(x + (c.xrow * T_ + 0) * D_ + c.d0));

  // t-loop unrolled by 2 so ping-pong buffer selection is static (immediate LDS offsets)
  for (int t = 0; t < T_; t += 2) {
    rnn_step(c, t,     buf0, buf1, xr, wreg);
    rnn_step(c, t + 1, buf1, buf0, xr, wreg);
  }
}

extern "C" void kernel_launch(void* const* d_in, const int* in_sizes, int n_in,
                              void* d_out, int out_size, void* d_ws, size_t ws_size,
                              hipStream_t stream) {
  const float* x    = (const float*)d_in[0];
  const float* h0   = (const float*)d_in[1];
  const float* W    = (const float*)d_in[2];
  const float* bias = (const float*)d_in[3];
  float* out = (float*)d_out;

  rnn_scan_wmma<<<dim3(B_ / 16), dim3(512), 0, stream>>>(x, h0, W, bias, out);
}
